// AdaptiveLSTMCell2_88115549045317
// MI455X (gfx1250) — matrix operand order repaired and checked by reading, not compiled, not run on hardware
//
#include <hip/hip_runtime.h>
#include <stdint.h>

// ---------------- problem dims ----------------
#define B_    128
#define IN_   256
#define H_    256
#define AH_   128
#define KBIG  4352            // 4096 (outer product) + 256 (bias rows)
#define NBIG  1024            // 4*H
#define NT_BIG 64             // 1024/16
#define KT_BIG 136            // 4352/32
#define W2F_PER ((size_t)NT_BIG * KT_BIG * 32 * 16)   // elems per problem = 4,456,448

#define CHUNK_KT 4            // k-tiles per TDM chunk (K=128 per chunk)
#define NCHUNK   (KT_BIG / CHUNK_KT)   // 34

// ---------------- workspace layout (bytes) ----------------
#define WS_W2F   ((size_t)0)                               // 2 * W2F_PER bf16 = 17,825,792 B
#define WS_IG    ((size_t)17825792)                        // 128*512 f32
#define WS_HG    (WS_IG + (size_t)128*512*4)               // 128*512 f32
#define WS_P     (WS_HG + (size_t)128*512*4)               // 2*128*4352 bf16
#define WS_C     (WS_P  + (size_t)2*128*4352*2)            // 2*128*1024 f32

// output layout: hy(128*256) | new_total_h(128*384) | new_total_c(128*384)
#define OUT_H    32768
#define OUT_C    81920

typedef __attribute__((ext_vector_type(16))) __bf16 bf16x16;
typedef __attribute__((ext_vector_type(8)))  __bf16 bf16x8;
typedef __attribute__((ext_vector_type(8)))  float  f32x8;
typedef __attribute__((ext_vector_type(4)))  unsigned int u32x4;
typedef __attribute__((ext_vector_type(4)))  int  i32x4;
typedef __attribute__((ext_vector_type(8)))  int  i32x8;

union FragA { bf16x16 v; bf16x8 h[2]; unsigned short u[16]; };
union FragB { bf16x16 v; unsigned short u[16]; };

// Tensor Data Mover availability (device pass on gfx1250 toolchains only)
#if defined(__has_builtin)
#if __has_builtin(__builtin_amdgcn_tensor_load_to_lds)
#define HAVE_TDM 1
#endif
#endif
#ifndef HAVE_TDM
#define HAVE_TDM 0
#endif

__device__ __forceinline__ unsigned short f2bf(float f) {
  union { float f; unsigned int u; } c; c.f = f;
  unsigned int u = c.u;
  u += 0x7fffu + ((u >> 16) & 1u);       // round-to-nearest-even
  return (unsigned short)(u >> 16);
}

__device__ __forceinline__ f32x8 wmma_bf16(bf16x16 a, bf16x16 b, f32x8 c) {
  return __builtin_amdgcn_wmma_f32_16x16x32_bf16(false, a, false, b, (short)0, c, false, false);
}

__device__ __forceinline__ float sigm(float x) { return 1.0f / (1.0f + __expf(-x)); }

template <int NT>
__device__ __forceinline__ float2 block_reduce2(float a, float b, float* r1, float* r2) {
  const int t = threadIdx.x;
  r1[t] = a; r2[t] = b; __syncthreads();
#pragma unroll
  for (int s = NT / 2; s > 0; s >>= 1) {
    if (t < s) { r1[t] += r1[t + s]; r2[t] += r2[t + s]; }
    __syncthreads();
  }
  float2 res; res.x = r1[0]; res.y = r2[0];
  __syncthreads();
  return res;
}

#if HAVE_TDM
// Issue one TDM 2-D tile load (data_size = 8 bytes). Tensor dims are set huge
// so no element is OOB-clamped; tiles are packed contiguously into LDS.
//   tile_d0 : tile width in 8-byte units (contiguous run)
//   tile_d1 : number of rows
//   stride0 : row stride in 8-byte units
__device__ __forceinline__ void tdm_load(uint32_t lds_off, const void* gptr,
                                         int tile_d0, int tile_d1, uint32_t stride0) {
  const uint32_t ga_lo = (uint32_t)__builtin_amdgcn_readfirstlane((int)(uint32_t)(uintptr_t)gptr);
  const uint32_t ga_hi = (uint32_t)__builtin_amdgcn_readfirstlane((int)(uint32_t)(((uint64_t)(uintptr_t)gptr) >> 32));
  lds_off = (uint32_t)__builtin_amdgcn_readfirstlane((int)lds_off);
  u32x4 g0;
  g0.x = 1u;                                       // count=1, user descriptor
  g0.y = lds_off;                                  // LDS byte address
  g0.z = ga_lo;                                    // global_addr[31:0]
  g0.w = (ga_hi & 0x01FFFFFFu) | (2u << 30);       // global_addr[56:32] | type=2
  i32x8 g1;
  g1[0] = 3 << 16;                                 // wg_mask=0, data_size=8B
  g1[1] = 0;                                       // tensor_dim0 = 0x40000000 (lo16=0)
  g1[2] = 0x4000;                                  // tensor_dim0 hi16; tensor_dim1 lo16=0
  g1[3] = 0x10 | (tile_d0 << 16);                  // tensor_dim1 hi16 (dim1=2^20); tile_dim0
  g1[4] = tile_d1;                                 // tile_dim1; tile_dim2=0
  g1[5] = (int)stride0;                            // tensor_dim0_stride[31:0]
  g1[6] = 0;                                       // stride0 hi; dim1_stride lo
  g1[7] = 0;
  i32x4 gz = {0, 0, 0, 0};
#if defined(__clang_major__) && __clang_major__ >= 23
  i32x8 gz8 = {0, 0, 0, 0, 0, 0, 0, 0};
  __builtin_amdgcn_tensor_load_to_lds(g0, g1, gz, gz, gz8, 0);
#else
  __builtin_amdgcn_tensor_load_to_lds(g0, g1, gz, gz, 0);
#endif
}
#endif // HAVE_TDM

// ============================================================================
// Kernel 1: reshape/convert big low-rank weights into WMMA-B fragment layout.
//   GEMM B-matrix: W2flat[K][N], K = i*16+k (K<4096) or 4096+i (bias rows).
//   Fragment tile (kt,nt): 32 lanes x 16 contiguous bf16; lane l holds column
//   N = nt*16 + (l&15), K = kt*32 + (l>=16 ? 8 : 0) + (e&7) + (e>=8 ? 16 : 0)  (e = 0..15).
//   For fixed nt, all kt are contiguous -> TDM copies are 1-D runs.
// ============================================================================
__global__ void k_prep(const float* __restrict__ w_ih2_W, const float* __restrict__ w_ih2_b,
                       const float* __restrict__ w_hh2_W, const float* __restrict__ w_hh2_b,
                       unsigned short* __restrict__ W2f) {
  size_t tid = (size_t)blockIdx.x * blockDim.x + threadIdx.x;
  const size_t per = W2F_PER;
  if (tid >= 2 * per) return;
  const int p = (tid >= per);
  size_t r = tid - (p ? per : 0);
  const int e    = (int)(r & 15);
  const int lane = (int)((r >> 4) & 31);
  size_t t2 = r >> 9;
  const int kt = (int)(t2 % KT_BIG);
  const int nt = (int)(t2 / KT_BIG);
  const int N = nt * 16 + (lane & 15);
  const int K = kt * 32 + ((lane >> 4) << 3) + (e & 7) + ((e >> 3) << 4);
  const float* W  = p ? w_hh2_W : w_ih2_W;
  const float* Bb = p ? w_hh2_b : w_ih2_b;
  float val;
  if (K < 4096) {
    const int i = K >> 4, k = K & 15;
    val = W[((size_t)i * 1024 + N) * 16 + k];
  } else {
    const int i = K - 4096;
    val = Bb[(size_t)i * 1024 + N];
  }
  W2f[tid] = f2bf(val);
}

// ============================================================================
// Kernel 2: adaptive-gate GEMMs with WMMA bf16 (f32 converted in flight).
//   which==0: ig_pre(128x512) = [input_|hx](128x512) @ a_w_ih^T   (K=512)
//   which==1: hg_pre(128x512) = a_h(128x128)         @ a_w_hh^T   (K=128)
// ============================================================================
__global__ void k_gemm_adapt(const float* __restrict__ input_, const float* __restrict__ total_h,
                             const float* __restrict__ a_w_ih, const float* __restrict__ a_w_hh,
                             float* __restrict__ ig_pre, float* __restrict__ hg_pre) {
  const int lane  = threadIdx.x & 31;
  const int nt    = blockIdx.x;
  const int mt    = blockIdx.y;
  const int which = blockIdx.z;
  const int half  = lane >> 4;
  const int mrow  = mt * 16 + (lane & 15);
  const int ncol  = nt * 16 + (lane & 15);
  f32x8 acc = {0.f, 0.f, 0.f, 0.f, 0.f, 0.f, 0.f, 0.f};

  const int Ktot = (which == 0) ? 512 : 128;
  for (int kb = 0; kb < Ktot; kb += 32) {
    const float* asrc;
    int k0;
    const float* bsrc;
    if (which == 0) {
      if (kb < 256) { asrc = input_ + (size_t)mrow * 256; k0 = kb; }
      else          { asrc = total_h + (size_t)mrow * 384; k0 = kb - 256; }   // hx
      bsrc = a_w_ih + (size_t)ncol * 512 + kb + half * 16;
    } else {
      asrc = total_h + (size_t)mrow * 384 + 256; k0 = kb;                      // a_h
      bsrc = a_w_hh + (size_t)ncol * 128 + kb + half * 16;
    }
    FragA a; FragB b;
    const int aoff = k0 + half * 16;
#pragma unroll
    for (int i = 0; i < 8; ++i) {
      a.u[i]     = f2bf(asrc[aoff + i]);
      a.u[8 + i] = f2bf(asrc[aoff + 8 + i]);
    }
#pragma unroll
    for (int i = 0; i < 16; ++i) b.u[i] = f2bf(bsrc[i]);
    acc = wmma_bf16(a.v, b.v, acc);
  }
  float* outp = (which == 0) ? ig_pre : hg_pre;
#pragma unroll
  for (int r = 0; r < 8; ++r)
    outp[(size_t)(mt * 16 + r + 8 * half) * 512 + nt * 16 + (lane & 15)] = acc[r];
}

// ============================================================================
// Kernel 3: per-batch-row adaptive cell (LNs, gates, tiny matvecs, P rows).
// ============================================================================
__global__ void k_adapt_cell(const float* __restrict__ ig_pre, const float* __restrict__ hg_pre,
                             const float* __restrict__ input_, const float* __restrict__ total_h,
                             const float* __restrict__ total_c,
                             const float* __restrict__ a_ln_i_w, const float* __restrict__ a_ln_i_b,
                             const float* __restrict__ a_ln_h_w, const float* __restrict__ a_ln_h_b,
                             const float* __restrict__ a_ln_c_w, const float* __restrict__ a_ln_c_b,
                             const float* __restrict__ w_ih1_W, const float* __restrict__ w_ih1_b,
                             const float* __restrict__ w_hh1_W, const float* __restrict__ w_hh1_b,
                             unsigned short* __restrict__ Pm, float* __restrict__ out) {
  __shared__ float r1[128], r2[128];
  __shared__ float g[512];
  __shared__ float ah[128];
  __shared__ float hbuf[32];
  const int b = blockIdx.x, t = threadIdx.x;

  float vi[4];
  float s = 0.f, s2 = 0.f;
#pragma unroll
  for (int q = 0; q < 4; ++q) { float x = ig_pre[(size_t)b * 512 + t * 4 + q]; vi[q] = x; s += x; s2 += x * x; }
  float2 red = block_reduce2<128>(s, s2, r1, r2);
  float mu = red.x * (1.f / 512.f);
  float rs = rsqrtf(red.y * (1.f / 512.f) - mu * mu + 1e-5f);
#pragma unroll
  for (int q = 0; q < 4; ++q) { int j = t * 4 + q; vi[q] = (vi[q] - mu) * rs * a_ln_i_w[j] + a_ln_i_b[j]; }

  float vh[4];
  s = 0.f; s2 = 0.f;
#pragma unroll
  for (int q = 0; q < 4; ++q) { float x = hg_pre[(size_t)b * 512 + t * 4 + q]; vh[q] = x; s += x; s2 += x * x; }
  red = block_reduce2<128>(s, s2, r1, r2);
  mu = red.x * (1.f / 512.f);
  rs = rsqrtf(red.y * (1.f / 512.f) - mu * mu + 1e-5f);
#pragma unroll
  for (int q = 0; q < 4; ++q) { int j = t * 4 + q; g[j] = vi[q] + (vh[q] - mu) * rs * a_ln_h_w[j] + a_ln_h_b[j]; }
  __syncthreads();

  const float gi = sigm(g[t]);
  const float gf = sigm(g[128 + t]);
  const float gc = tanhf(g[256 + t]);
  const float go = sigm(g[384 + t]);
  const float a_c = total_c[(size_t)b * 384 + 256 + t];
  const float pre = gf * a_c + gi * gc;
  red = block_reduce2<128>(pre, pre * pre, r1, r2);
  mu = red.x * (1.f / 128.f);
  rs = rsqrtf(red.y * (1.f / 128.f) - mu * mu + 1e-5f);
  const float acn = (pre - mu) * rs * a_ln_c_w[t] + a_ln_c_b[t];
  const float ahn = go * tanhf(acn);
  out[OUT_H + (size_t)b * 384 + 256 + t] = ahn;
  out[OUT_C + (size_t)b * 384 + 256 + t] = acn;
  ah[t] = ahn;
  __syncthreads();

  if (t < 32) {
    const int j = t & 15;
    const float* wv = (t < 16) ? (w_ih1_W + (size_t)j * 128) : (w_hh1_W + (size_t)j * 128);
    float acc = (t < 16) ? w_ih1_b[j] : w_hh1_b[j];
    for (int k = 0; k < 128; ++k) acc += ah[k] * wv[k];
    hbuf[t] = acc;
  }
  __syncthreads();

  unsigned short* P0 = Pm + (size_t)b * KBIG;
  unsigned short* P1 = Pm + (size_t)128 * KBIG + (size_t)b * KBIG;
  for (int idx = t; idx < KBIG; idx += 128) {
    float v0, v1;
    if (idx < 4096) {
      const int i = idx >> 4, k = idx & 15;
      v0 = input_[(size_t)b * 256 + i] * hbuf[k];
      v1 = total_h[(size_t)b * 384 + i] * hbuf[16 + k];     // hx
    } else {
      const int i = idx - 4096;
      v0 = input_[(size_t)b * 256 + i];
      v1 = total_h[(size_t)b * 384 + i];
    }
    P0[idx] = f2bf(v0);
    P1[idx] = f2bf(v1);
  }
}

// ============================================================================
// Kernel 4: dominant GEMMs: C[p](128x1024) = P[p](128x4352) @ W2f[p].
//   Block = 256 thr = 8 waves; block tile 32(M) x 128(N); grid (8,4,2).
//   TDM double-buffered LDS staging: each wave DMAs its own B strip (4 KB
//   contiguous) + 4 A rows per chunk; s_wait_tensorcnt(2) keeps two chunks
//   in flight; compute reads fragments from LDS (ds_load_b128).
// ============================================================================
__global__ void __launch_bounds__(256) k_gemm_main(const unsigned short* __restrict__ Pm,
                                                   const unsigned short* __restrict__ W2f,
                                                   float* __restrict__ Cm) {
  __shared__ __align__(16) unsigned short a_lds[2][32][CHUNK_KT * 32];       // 16 KB
  __shared__ __align__(16) unsigned short b_lds[2][8][CHUNK_KT][32][16];     // 64 KB

  const int p = blockIdx.z;
  const unsigned short* Pp = Pm + (size_t)p * 128 * KBIG;
  const unsigned short* Wp = W2f + (size_t)p * W2F_PER;
  float* Cp = Cm + (size_t)p * 128 * 1024;

  const int lane = threadIdx.x & 31;
  const int wave = __builtin_amdgcn_readfirstlane((int)(threadIdx.x >> 5));  // 0..7, uniform
  const int half = lane >> 4;

  // compute assignment
  const int mloc = (wave >> 2) * 16 + (lane & 15);      // local row 0..31
  const int ntl0 = (wave & 3) * 2;                      // local B strip pair
  const int nbase = blockIdx.x * 128 + (wave & 3) * 32;

  // copy assignment (per wave)
  const int nt_g  = blockIdx.x * 8 + wave;              // B strip this wave DMAs
  const int arow0 = blockIdx.y * 32 + wave * 4;         // A rows this wave DMAs

  f32x8 acc0 = {0.f, 0.f, 0.f, 0.f, 0.f, 0.f, 0.f, 0.f};
  f32x8 acc1 = {0.f, 0.f, 0.f, 0.f, 0.f, 0.f, 0.f, 0.f};

#if HAVE_TDM
  auto issue = [&](int c, int buf) {
    // B strip: one contiguous run of CHUNK_KT k-tiles = 4 KB
    tdm_load((uint32_t)(uintptr_t)&b_lds[buf][wave][0][0][0],
             Wp + ((size_t)nt_g * KT_BIG + (size_t)c * CHUNK_KT) * 512,
             /*tile_d0=*/CHUNK_KT * 128, /*tile_d1=*/1, /*stride0=*/CHUNK_KT * 128);
    // A rows: 4 rows x 256 B, row stride 8704 B (= 1088 x 8B)
    tdm_load((uint32_t)(uintptr_t)&a_lds[buf][wave * 4][0],
             Pp + (size_t)arow0 * KBIG + (size_t)c * (CHUNK_KT * 32),
             /*tile_d0=*/CHUNK_KT * 8, /*tile_d1=*/4, /*stride0=*/KBIG / 4);
  };
  issue(0, 0);
  issue(1, 1);
  for (int c = 0; c < NCHUNK; ++c) {
    if (c + 1 < NCHUNK) __builtin_amdgcn_s_wait_tensorcnt(2);   // chunk c arrived
    else                __builtin_amdgcn_s_wait_tensorcnt(0);
    __syncthreads();                                            // all waves' slices visible
    const int buf = c & 1;
#pragma unroll
    for (int ktl = 0; ktl < CHUNK_KT; ++ktl) {
      FragA a;
      a.h[0] = *(const bf16x8*)&a_lds[buf][mloc][ktl * 32 + half * 8];
      a.h[1] = *(const bf16x8*)&a_lds[buf][mloc][ktl * 32 + 16 + half * 8];
      FragB b0, b1;
      b0.v = *(const bf16x16*)&b_lds[buf][ntl0][ktl][lane][0];
      b1.v = *(const bf16x16*)&b_lds[buf][ntl0 + 1][ktl][lane][0];
      acc0 = wmma_bf16(a.v, b0.v, acc0);
      acc1 = wmma_bf16(a.v, b1.v, acc1);
    }
    __syncthreads();                                            // buf free for refill
    if (c + 2 < NCHUNK) issue(c + 2, buf);
  }
#else
  // Fallback: direct-from-global compute (no TDM on this toolchain)
  const int mrow_g = blockIdx.y * 32 + mloc;
  const unsigned short* arow = Pp + (size_t)mrow_g * KBIG;
  const int nt0 = nbase >> 4;
  const unsigned short* bptr0 = Wp + ((size_t)nt0 * KT_BIG * 32) * 16 + (size_t)lane * 16;
  const unsigned short* bptr1 = Wp + ((size_t)(nt0 + 1) * KT_BIG * 32) * 16 + (size_t)lane * 16;
  for (int kt = 0; kt < KT_BIG; ++kt) {
    const int kb = kt * 32;
    FragA a;
    a.h[0] = *(const bf16x8*)(arow + kb + half * 8);
    a.h[1] = *(const bf16x8*)(arow + kb + 16 + half * 8);
    FragB b0, b1;
    b0.v = *(const bf16x16*)(bptr0 + (size_t)kt * 32 * 16);
    b1.v = *(const bf16x16*)(bptr1 + (size_t)kt * 32 * 16);
    acc0 = wmma_bf16(a.v, b0.v, acc0);
    acc1 = wmma_bf16(a.v, b1.v, acc1);
  }
#endif

#pragma unroll
  for (int r = 0; r < 8; ++r) {
    const int row = blockIdx.y * 32 + (wave >> 2) * 16 + r + 8 * half;
    Cp[(size_t)row * 1024 + nbase + (lane & 15)]      = acc0[r];
    Cp[(size_t)row * 1024 + nbase + 16 + (lane & 15)] = acc1[r];
  }
}

// ============================================================================
// Kernel 5: final LSTM pointwise: LN(igates)+LN(hgates) -> gates -> cy, hy.
// ============================================================================
__global__ void k_final(const float* __restrict__ Cm, const float* __restrict__ total_c,
                        const float* __restrict__ ln_i_w, const float* __restrict__ ln_i_b,
                        const float* __restrict__ ln_h_w, const float* __restrict__ ln_h_b,
                        float* __restrict__ out) {
  __shared__ float r1[256], r2[256];
  __shared__ float g[1024];
  const int b = blockIdx.x, t = threadIdx.x;
  const float* igp = Cm + (size_t)b * 1024;
  const float* hgp = Cm + (size_t)128 * 1024 + (size_t)b * 1024;

  float vi[4];
  float s = 0.f, s2 = 0.f;
#pragma unroll
  for (int q = 0; q < 4; ++q) { float x = igp[t * 4 + q]; vi[q] = x; s += x; s2 += x * x; }
  float2 red = block_reduce2<256>(s, s2, r1, r2);
  float mu = red.x * (1.f / 1024.f);
  float rs = rsqrtf(red.y * (1.f / 1024.f) - mu * mu + 1e-5f);
#pragma unroll
  for (int q = 0; q < 4; ++q) { int j = t * 4 + q; vi[q] = (vi[q] - mu) * rs * ln_i_w[j] + ln_i_b[j]; }

  float vh[4];
  s = 0.f; s2 = 0.f;
#pragma unroll
  for (int q = 0; q < 4; ++q) { float x = hgp[t * 4 + q]; vh[q] = x; s += x; s2 += x * x; }
  red = block_reduce2<256>(s, s2, r1, r2);
  mu = red.x * (1.f / 1024.f);
  rs = rsqrtf(red.y * (1.f / 1024.f) - mu * mu + 1e-5f);
#pragma unroll
  for (int q = 0; q < 4; ++q) { int j = t * 4 + q; g[j] = vi[q] + (vh[q] - mu) * rs * ln_h_w[j] + ln_h_b[j]; }
  __syncthreads();

  const float i_ = sigm(g[t]);
  const float f_ = sigm(g[256 + t]);
  const float c_ = tanhf(g[512 + t]);
  const float o_ = sigm(g[768 + t]);
  const float cx = total_c[(size_t)b * 384 + t];
  const float cy = f_ * cx + i_ * c_;
  const float hy = o_ * tanhf(cy);
  out[(size_t)b * 256 + t] = hy;
  out[OUT_H + (size_t)b * 384 + t] = hy;
  out[OUT_C + (size_t)b * 384 + t] = cy;
}

// ============================================================================
extern "C" void kernel_launch(void* const* d_in, const int* in_sizes, int n_in,
                              void* d_out, int out_size, void* d_ws, size_t ws_size,
                              hipStream_t stream) {
  const float* input_   = (const float*)d_in[0];
  const float* total_h  = (const float*)d_in[1];
  const float* total_c  = (const float*)d_in[2];
  const float* a_w_ih   = (const float*)d_in[3];
  const float* a_w_hh   = (const float*)d_in[4];
  const float* a_ln_i_w = (const float*)d_in[5];
  const float* a_ln_i_b = (const float*)d_in[6];
  const float* a_ln_h_w = (const float*)d_in[7];
  const float* a_ln_h_b = (const float*)d_in[8];
  const float* a_ln_c_w = (const float*)d_in[9];
  const float* a_ln_c_b = (const float*)d_in[10];
  const float* w_ih1_W  = (const float*)d_in[11];
  const float* w_ih1_b  = (const float*)d_in[12];
  const float* w_ih2_W  = (const float*)d_in[13];
  const float* w_ih2_b  = (const float*)d_in[14];
  const float* w_hh1_W  = (const float*)d_in[15];
  const float* w_hh1_b  = (const float*)d_in[16];
  const float* w_hh2_W  = (const float*)d_in[17];
  const float* w_hh2_b  = (const float*)d_in[18];
  const float* ln_i_w   = (const float*)d_in[19];
  const float* ln_i_b   = (const float*)d_in[20];
  const float* ln_h_w   = (const float*)d_in[21];
  const float* ln_h_b   = (const float*)d_in[22];

  char* ws = (char*)d_ws;
  unsigned short* W2f = (unsigned short*)(ws + WS_W2F);
  float*          igp = (float*)(ws + WS_IG);
  float*          hgp = (float*)(ws + WS_HG);
  unsigned short* Pm  = (unsigned short*)(ws + WS_P);
  float*          Cm  = (float*)(ws + WS_C);
  float*          out = (float*)d_out;

  {
    const size_t total = 2 * W2F_PER;
    const int threads = 256;
    const int blocks = (int)((total + threads - 1) / threads);
    k_prep<<<blocks, threads, 0, stream>>>(w_ih2_W, w_ih2_b, w_hh2_W, w_hh2_b, W2f);
  }
  k_gemm_adapt<<<dim3(32, 8, 2), 32, 0, stream>>>(input_, total_h, a_w_ih, a_w_hh, igp, hgp);
  k_adapt_cell<<<128, 128, 0, stream>>>(igp, hgp, input_, total_h, total_c,
                                        a_ln_i_w, a_ln_i_b, a_ln_h_w, a_ln_h_b,
                                        a_ln_c_w, a_ln_c_b,
                                        w_ih1_W, w_ih1_b, w_hh1_W, w_hh1_b, Pm, out);
  k_gemm_main<<<dim3(8, 4, 2), 256, 0, stream>>>(Pm, W2f, Cm);
  k_final<<<128, 256, 0, stream>>>(Cm, total_c, ln_i_w, ln_i_b, ln_h_w, ln_h_b, out);
}
